// SharedExpertMoE_37237366456750
// MI455X (gfx1250) — matrix operand order repaired and checked by reading, not compiled, not run on hardware
//
#include <hip/hip_runtime.h>
#include <hip/hip_bf16.h>
#include <math.h>

// ---------------- problem constants (match reference) ----------------
#define DIMX     1024
#define NEXP     8
#define NSH      2
#define EXP_HID  4096      // routed SwiGLU hidden
#define SH_HID   2048      // shared SwiGLU hidden
#define TOK      8192      // B*S = 4*2048

typedef __bf16 bf16;
typedef __attribute__((ext_vector_type(16))) __bf16 v16bf;
typedef __attribute__((ext_vector_type(8)))  float  v8f;
typedef __attribute__((ext_vector_type(4)))  __bf16 bf16x4;

// ---------------- WMMA helper (CDNA5 wave32, 16x16x32 bf16) ----------------
__device__ __forceinline__ v8f wmma_bf16(v16bf a, v16bf b, v8f c) {
  return __builtin_amdgcn_wmma_f32_16x16x32_bf16(
      /*neg_a=*/false, a, /*neg_b=*/false, b,
      /*c_mod=*/(short)0, c, /*reuse_a=*/false, /*reuse_b=*/false);
}

// A fragment 16x32 bf16 from LDS tile smA[128][64] (row stride 64, k contiguous).
// ISA: M = lane%16 ; lanes<16 hold K {0..7,16..23}, lanes>=16 hold K {8..15,24..31}
__device__ __forceinline__ v16bf frag_a64(const bf16* smA, int m0, int ks, int lane) {
  union { v16bf v; float4 q[2]; } u;
  const bf16* p = smA + (m0 + (lane & 15)) * 64 + ks + ((lane & 16) ? 8 : 0);
  u.q[0] = *(const float4*)(p);
  u.q[1] = *(const float4*)(p + 16);
  return u.v;
}

// B fragment 32x16 bf16 from LDS tile smB[N][64] (transposed, k contiguous per col).
// ISA: N = lane%16 ; lanes<16 hold K {0..7,16..23}, lanes>=16 hold K {8..15,24..31}
__device__ __forceinline__ v16bf frag_b64(const bf16* smB, int n0, int ks, int lane) {
  union { v16bf v; float4 q[2]; } u;
  const bf16* p = smB + (n0 + (lane & 15)) * 64 + ks + ((lane & 16) ? 8 : 0);
  u.q[0] = *(const float4*)(p);
  u.q[1] = *(const float4*)(p + 16);
  return u.v;
}

__device__ __forceinline__ uint32_t lds_off(const void* p) {
  return (uint32_t)(uintptr_t)p;   // low 32 bits of flat shared addr = LDS offset
}

__device__ __forceinline__ uint32_t pack_bf16(float a, float b) {
  union { bf16 h[2]; uint32_t u; } p;     // -> v_cvt_pk_bf16_f32
  p.h[0] = (bf16)a; p.h[1] = (bf16)b;
  return p.u;
}

// ---- async A staging: 128x64 bf16 tile = 1024 x 16B transfers, 4 per thread ----
struct AsyncA {
  uint64_t ga[4];    // global byte address of this thread's 16B slice (k0 = 0)
  uint32_t lo[4];    // LDS byte offset within one A buffer
  bool     en[4];    // row valid
};

__device__ __forceinline__ AsyncA make_asyncA(const bf16* A, const int* list,
                                              int m0, int M, int K, int tid) {
  AsyncA s;
#pragma unroll
  for (int h = 0; h < 4; ++h) {
    int t   = tid + h * 256;       // 0..1023
    int r   = t >> 3;              // row in tile
    int seg = t & 7;               // 16B segment within 128B row slice
    int row = m0 + r;
    s.en[h] = row < M;
    int g   = s.en[h] ? (list ? list[row] : row) : 0;
    s.ga[h] = (uint64_t)(uintptr_t)(A + (size_t)g * K) + (uint64_t)seg * 16;
    s.lo[h] = (uint32_t)(r * 128 + seg * 16);
  }
  return s;
}

__device__ __forceinline__ void issue_asyncA(const AsyncA& s, uint32_t ldsbase, int k0) {
#pragma unroll
  for (int h = 0; h < 4; ++h) {
    if (s.en[h]) {
      uint32_t dst = ldsbase + s.lo[h];
      uint64_t src = s.ga[h] + (uint64_t)k0 * 2;   // k0 bf16 elements -> bytes
      asm volatile("global_load_async_to_lds_b128 %0, %1, off"
                   :: "v"(dst), "v"(src) : "memory");
    }
  }
}

__device__ __forceinline__ void wait_async0() {
  asm volatile("s_wait_asynccnt 0" ::: "memory");
}

// zero the ragged tail rows of both A buffers once (async loads skip them)
__device__ __forceinline__ void zero_tailA(bf16* a0, bf16* a1, int tail, int tid) {
  if (tail < 128) {
    for (int i = tid; i < (128 - tail) * 16; i += 256) {
      int r = tail + (i >> 4), c = (i & 15) << 2;
      bf16x4 z = {};
      *(bf16x4*)(a0 + r * 64 + c) = z;
      *(bf16x4*)(a1 + r * 64 + c) = z;
    }
  }
}

// ---- B staging: 64k x BN(n) f32 -> transposed bf16 smB[BN][64] ----
// micro-tile per chunk: 2(k) x 4(n); load two float4s, store 4 packed b32.
template <int BN, int MPT>
__device__ __forceinline__ void loadB(float4* f, const float* __restrict__ W,
                                      int Nw, int n0, int k0, int tid) {
#pragma unroll
  for (int h = 0; h < MPT; ++h) {
    int c  = tid + h * 256;
    int kk = (c / (BN / 4)) * 2;
    int n4 = (c % (BN / 4)) * 4;
    f[2 * h]     = *(const float4*)(W + (size_t)(k0 + kk)     * Nw + n0 + n4);
    f[2 * h + 1] = *(const float4*)(W + (size_t)(k0 + kk + 1) * Nw + n0 + n4);
  }
}

template <int BN, int MPT>
__device__ __forceinline__ void storeB(bf16* smB, const float4* f, int tid) {
#pragma unroll
  for (int h = 0; h < MPT; ++h) {
    int c  = tid + h * 256;
    int kk = (c / (BN / 4)) * 2;
    int n4 = (c % (BN / 4)) * 4;
    const float4& a = f[2 * h];
    const float4& b = f[2 * h + 1];
    *(uint32_t*)(smB + (n4 + 0) * 64 + kk) = pack_bf16(a.x, b.x);
    *(uint32_t*)(smB + (n4 + 1) * 64 + kk) = pack_bf16(a.y, b.y);
    *(uint32_t*)(smB + (n4 + 2) * 64 + kk) = pack_bf16(a.z, b.z);
    *(uint32_t*)(smB + (n4 + 3) * 64 + kk) = pack_bf16(a.w, b.w);
  }
}

// ---------------- kernel: zero output + counters ----------------
__global__ __launch_bounds__(256) void zero_kernel(float* __restrict__ out,
                                                   int* __restrict__ cnt,
                                                   int* __restrict__ cnt1) {
  size_t base = ((size_t)blockIdx.x * 256 + threadIdx.x) * 8;
  float4 z = {0.f, 0.f, 0.f, 0.f};
  *(float4*)(out + base)     = z;
  *(float4*)(out + base + 4) = z;
  if (blockIdx.x == 0 && threadIdx.x < NEXP) { cnt[threadIdx.x] = 0; cnt1[threadIdx.x] = 0; }
}

// ---------------- kernel: fp32 gating, top-2 softmax, list build ----------------
__global__ __launch_bounds__(256) void gate_kernel(const float* __restrict__ x,
                                                   const float* __restrict__ gw,
                                                   int* __restrict__ toklist,
                                                   float* __restrict__ gatelist,
                                                   int* __restrict__ cnt,
                                                   int* __restrict__ cnt1) {
  __shared__ float sgw[DIMX * NEXP];
  for (int i = threadIdx.x; i < DIMX * NEXP; i += 256) sgw[i] = gw[i];
  __syncthreads();
  int wid = threadIdx.x >> 5, lane = threadIdx.x & 31;
  int t = blockIdx.x * 8 + wid;                 // one wave per token
  float acc[NEXP];
#pragma unroll
  for (int e = 0; e < NEXP; ++e) acc[e] = 0.f;
  const float* xr = x + (size_t)t * DIMX;
  for (int d = lane; d < DIMX; d += 32) {
    float xv = xr[d];
    const float* g = sgw + d * NEXP;
#pragma unroll
    for (int e = 0; e < NEXP; ++e) acc[e] += xv * g[e];
  }
#pragma unroll
  for (int e = 0; e < NEXP; ++e)
    for (int off = 16; off; off >>= 1) acc[e] += __shfl_xor(acc[e], off, 32);
  if (lane == 0) {
    int i0 = 0; float v0 = acc[0];
    for (int e = 1; e < NEXP; ++e) if (acc[e] > v0) { v0 = acc[e]; i0 = e; }
    int i1 = (i0 == 0) ? 1 : 0; float v1 = acc[i1];
    for (int e = 0; e < NEXP; ++e) if (e != i0 && acc[e] > v1) { v1 = acc[e]; i1 = e; }
    float e1 = __expf(v1 - v0);                 // v1 <= v0, stable softmax
    float g0 = 1.f / (1.f + e1), g1 = e1 / (1.f + e1);
    int p0 = atomicAdd(&cnt[i0], 1);
    toklist[i0 * TOK + p0] = t; gatelist[i0 * TOK + p0] = g0;
    int p1 = atomicAdd(&cnt[i1], 1);
    toklist[i1 * TOK + p1] = t; gatelist[i1 * TOK + p1] = g1;
    atomicAdd(&cnt1[i0], 1);
  }
}

// ---------------- kernel: load-balance loss ----------------
__global__ void lb_kernel(const int* __restrict__ cnt1, float* __restrict__ loss) {
  if (threadIdx.x == 0 && blockIdx.x == 0) {
    float tot = 0.f;
    for (int e = 0; e < NEXP; ++e) tot += (float)cnt1[e];
    float s = 0.f;
    for (int e = 0; e < NEXP; ++e) {
      float d = (float)cnt1[e] / tot - 1.f / NEXP;
      s += d * d;
    }
    loss[0] = s / NEXP;
  }
}

// ---------------- kernel: x f32 -> bf16 ----------------
__global__ __launch_bounds__(256) void cvt_kernel(const float* __restrict__ x,
                                                  bf16* __restrict__ X16) {
  size_t i = ((size_t)blockIdx.x * 256 + threadIdx.x) * 4;
  float4 f = *(const float4*)(x + i);
  bf16x4 v = {(bf16)f.x, (bf16)f.y, (bf16)f.z, (bf16)f.w};
  *(bf16x4*)(X16 + i) = v;
}

// ------- kernel: C = swiglu(A @ W1)  [128Mx64N, K-step 64, double-buffered] -------
// W1 is [K x 2*Hdim]; a-half cols [n0..), g-half cols [Hdim+n0..). Output bf16 [rows x Hdim].
__global__ __launch_bounds__(256)
void gemm_swiglu(const bf16* __restrict__ A, const float* __restrict__ W1,
                 bf16* __restrict__ Hout, const int* __restrict__ list,
                 const int* __restrict__ cntp, int e, int Mtot, int K, int Hdim) {
  __shared__ bf16 smA[2][128 * 64];
  __shared__ bf16 smBa[2][64 * 64];
  __shared__ bf16 smBg[2][64 * 64];
  int M = cntp ? cntp[e] : Mtot;
  int m0 = blockIdx.y * 128;
  if (m0 >= M) return;                           // block-uniform: EXEC full for WMMA
  int n0 = blockIdx.x * 64;
  int tid = threadIdx.x, lane = tid & 31, wid = tid >> 5;
  int wm = (wid >> 1) * 32;                      // 4 waves over M
  int wn = (wid & 1) * 32;                       // 2 waves over N
  int Nw = 2 * Hdim;

  AsyncA aa = make_asyncA(A, list, m0, M, K, tid);
  zero_tailA(smA[0], smA[1], M - m0, tid);
  uint32_t ldsA[2] = {lds_off(smA[0]), lds_off(smA[1])};

  v8f acc_a[2][2] = {}; v8f acc_g[2][2] = {};
  float4 fba[4], fbg[4];

  issue_asyncA(aa, ldsA[0], 0);
  loadB<64, 2>(fba, W1, Nw, n0, 0, tid);
  loadB<64, 2>(fbg, W1, Nw, n0 + Hdim, 0, tid);
  storeB<64, 2>(smBa[0], fba, tid);
  storeB<64, 2>(smBg[0], fbg, tid);
  wait_async0();
  __syncthreads();

  int nk = K / 64;
  for (int i = 0; i < nk; ++i) {
    int cur = i & 1, nxt = cur ^ 1;
    if (i + 1 < nk) {
      issue_asyncA(aa, ldsA[nxt], (i + 1) * 64);
      loadB<64, 2>(fba, W1, Nw, n0, (i + 1) * 64, tid);
      loadB<64, 2>(fbg, W1, Nw, n0 + Hdim, (i + 1) * 64, tid);
    }
#pragma unroll
    for (int ks = 0; ks < 64; ks += 32) {
      v16bf af[2], b0[2], b1[2];
      af[0] = frag_a64(smA[cur], wm, ks, lane);       af[1] = frag_a64(smA[cur], wm + 16, ks, lane);
      b0[0] = frag_b64(smBa[cur], wn, ks, lane);      b0[1] = frag_b64(smBa[cur], wn + 16, ks, lane);
      b1[0] = frag_b64(smBg[cur], wn, ks, lane);      b1[1] = frag_b64(smBg[cur], wn + 16, ks, lane);
#pragma unroll
      for (int ii = 0; ii < 2; ++ii)
#pragma unroll
        for (int j = 0; j < 2; ++j) {
          acc_a[ii][j] = wmma_bf16(af[ii], b0[j], acc_a[ii][j]);
          acc_g[ii][j] = wmma_bf16(af[ii], b1[j], acc_g[ii][j]);
        }
    }
    if (i + 1 < nk) {
      storeB<64, 2>(smBa[nxt], fba, tid);
      storeB<64, 2>(smBg[nxt], fbg, tid);
    }
    wait_async0();
    __syncthreads();
  }
#pragma unroll
  for (int ii = 0; ii < 2; ++ii)
#pragma unroll
    for (int j = 0; j < 2; ++j)
#pragma unroll
      for (int r = 0; r < 8; ++r) {
        int row = m0 + wm + ii * 16 + r + ((lane & 16) ? 8 : 0);
        if (row < M) {
          int col = n0 + wn + j * 16 + (lane & 15);
          float a = acc_a[ii][j][r], g = acc_g[ii][j][r];
          float s = a / (1.f + __expf(-a));      // silu
          Hout[(size_t)row * Hdim + col] = (bf16)(s * g);
        }
      }
}

// ------- kernel: Out(bf16) = A @ W  [128Mx128N, K-step 64, double-buffered] -------
__global__ __launch_bounds__(256)
void gemm_mid(const bf16* __restrict__ A, const float* __restrict__ W,
              bf16* __restrict__ Out, const int* __restrict__ cntp, int e,
              int K, int N) {
  __shared__ bf16 smA[2][128 * 64];
  __shared__ bf16 smB[2][128 * 64];
  int M = cntp[e];
  int m0 = blockIdx.y * 128;
  if (m0 >= M) return;
  int n0 = blockIdx.x * 128;
  int tid = threadIdx.x, lane = tid & 31, wid = tid >> 5;
  int wm = (wid >> 1) * 32;
  int wn = (wid & 1) * 64;

  AsyncA aa = make_asyncA(A, nullptr, m0, M, K, tid);
  zero_tailA(smA[0], smA[1], M - m0, tid);
  uint32_t ldsA[2] = {lds_off(smA[0]), lds_off(smA[1])};

  v8f acc[2][4] = {};
  float4 fb[8];

  issue_asyncA(aa, ldsA[0], 0);
  loadB<128, 4>(fb, W, N, n0, 0, tid);
  storeB<128, 4>(smB[0], fb, tid);
  wait_async0();
  __syncthreads();

  int nk = K / 64;
  for (int i = 0; i < nk; ++i) {
    int cur = i & 1, nxt = cur ^ 1;
    if (i + 1 < nk) {
      issue_asyncA(aa, ldsA[nxt], (i + 1) * 64);
      loadB<128, 4>(fb, W, N, n0, (i + 1) * 64, tid);
    }
#pragma unroll
    for (int ks = 0; ks < 64; ks += 32) {
      v16bf af[2], bfr[4];
      af[0] = frag_a64(smA[cur], wm, ks, lane); af[1] = frag_a64(smA[cur], wm + 16, ks, lane);
#pragma unroll
      for (int j = 0; j < 4; ++j) bfr[j] = frag_b64(smB[cur], wn + j * 16, ks, lane);
#pragma unroll
      for (int ii = 0; ii < 2; ++ii)
#pragma unroll
        for (int j = 0; j < 4; ++j) acc[ii][j] = wmma_bf16(af[ii], bfr[j], acc[ii][j]);
    }
    if (i + 1 < nk) storeB<128, 4>(smB[nxt], fb, tid);
    wait_async0();
    __syncthreads();
  }
#pragma unroll
  for (int ii = 0; ii < 2; ++ii)
#pragma unroll
    for (int j = 0; j < 4; ++j)
#pragma unroll
      for (int r = 0; r < 8; ++r) {
        int row = m0 + wm + ii * 16 + r + ((lane & 16) ? 8 : 0);
        if (row < M) {
          int col = n0 + wn + j * 16 + (lane & 15);
          Out[(size_t)row * N + col] = (bf16)acc[ii][j][r];
        }
      }
}

// --- kernel: Out(f32) += w * (A @ W + bias), optional row scatter [128x128, K-step 64] ---
__global__ __launch_bounds__(256)
void gemm_acc(const bf16* __restrict__ A, const float* __restrict__ W,
              float* __restrict__ Out, const int* __restrict__ list,
              const float* __restrict__ gates, const float* __restrict__ bias,
              const int* __restrict__ cntp, int e, int Mtot, int K, int N,
              float scale) {
  __shared__ bf16 smA[2][128 * 64];
  __shared__ bf16 smB[2][128 * 64];
  int M = cntp ? cntp[e] : Mtot;
  int m0 = blockIdx.y * 128;
  if (m0 >= M) return;
  int n0 = blockIdx.x * 128;
  int tid = threadIdx.x, lane = tid & 31, wid = tid >> 5;
  int wm = (wid >> 1) * 32;
  int wn = (wid & 1) * 64;

  AsyncA aa = make_asyncA(A, nullptr, m0, M, K, tid);
  zero_tailA(smA[0], smA[1], M - m0, tid);
  uint32_t ldsA[2] = {lds_off(smA[0]), lds_off(smA[1])};

  v8f acc[2][4] = {};
  float4 fb[8];

  issue_asyncA(aa, ldsA[0], 0);
  loadB<128, 4>(fb, W, N, n0, 0, tid);
  storeB<128, 4>(smB[0], fb, tid);
  wait_async0();
  __syncthreads();

  int nk = K / 64;
  for (int i = 0; i < nk; ++i) {
    int cur = i & 1, nxt = cur ^ 1;
    if (i + 1 < nk) {
      issue_asyncA(aa, ldsA[nxt], (i + 1) * 64);
      loadB<128, 4>(fb, W, N, n0, (i + 1) * 64, tid);
    }
#pragma unroll
    for (int ks = 0; ks < 64; ks += 32) {
      v16bf af[2], bfr[4];
      af[0] = frag_a64(smA[cur], wm, ks, lane); af[1] = frag_a64(smA[cur], wm + 16, ks, lane);
#pragma unroll
      for (int j = 0; j < 4; ++j) bfr[j] = frag_b64(smB[cur], wn + j * 16, ks, lane);
#pragma unroll
      for (int ii = 0; ii < 2; ++ii)
#pragma unroll
        for (int j = 0; j < 4; ++j) acc[ii][j] = wmma_bf16(af[ii], bfr[j], acc[ii][j]);
    }
    if (i + 1 < nk) storeB<128, 4>(smB[nxt], fb, tid);
    wait_async0();
    __syncthreads();
  }
#pragma unroll
  for (int ii = 0; ii < 2; ++ii)
#pragma unroll
    for (int j = 0; j < 4; ++j)
#pragma unroll
      for (int r = 0; r < 8; ++r) {
        int rloc = m0 + wm + ii * 16 + r + ((lane & 16) ? 8 : 0);
        if (rloc < M) {
          int orow = list ? list[rloc] : rloc;
          float wgt = gates ? gates[rloc] : scale;
          int col = n0 + wn + j * 16 + (lane & 15);
          float v = acc[ii][j][r] + (bias ? bias[col] : 0.f);
          Out[(size_t)orow * N + col] += wgt * v;   // rows unique within launch
        }
      }
}

// ---------------- host orchestration ----------------
extern "C" void kernel_launch(void* const* d_in, const int* in_sizes, int n_in,
                              void* d_out, int out_size, void* d_ws, size_t ws_size,
                              hipStream_t stream) {
  const float* x      = (const float*)d_in[0];
  const float* gate_w = (const float*)d_in[1];
  const float* exp_w1 = (const float*)d_in[2];
  const float* exp_w2 = (const float*)d_in[3];
  const float* exp_wo = (const float*)d_in[4];
  const float* exp_bo = (const float*)d_in[5];
  const float* sh_w1  = (const float*)d_in[6];
  const float* sh_w2  = (const float*)d_in[7];
  float* out = (float*)d_out;

  char* ws = (char*)d_ws;
  size_t off = 0;
  auto alloc = [&](size_t bytes) -> void* {
    void* p = ws + off;
    off = (off + bytes + 255) & ~(size_t)255;
    return p;
  };
  bf16*  X16      = (bf16*)alloc((size_t)TOK * DIMX * 2);
  bf16*  Hbuf     = (bf16*)alloc((size_t)TOK * EXP_HID * 2);
  bf16*  MID      = (bf16*)alloc((size_t)TOK * DIMX * 2);
  int*   toklist  = (int*)alloc((size_t)NEXP * TOK * 4);
  float* gatelist = (float*)alloc((size_t)NEXP * TOK * 4);
  int*   cnt      = (int*)alloc(64);
  int*   cnt1     = (int*)alloc(64);

  // 1) zero output accumulator + counters
  zero_kernel<<<(TOK * DIMX) / (256 * 8), 256, 0, stream>>>(out, cnt, cnt1);
  // 2) gating (fp32 exact), list build
  gate_kernel<<<TOK / 8, 256, 0, stream>>>(x, gate_w, toklist, gatelist, cnt, cnt1);
  // 3) load-balance loss -> d_out[TOK*DIMX]
  lb_kernel<<<1, 32, 0, stream>>>(cnt1, out + (size_t)TOK * DIMX);
  // 4) x -> bf16
  cvt_kernel<<<(TOK * DIMX) / (256 * 4), 256, 0, stream>>>(x, X16);

  // 5) shared experts: out += 0.5 * swiglu(x@sh_w1[s]) @ sh_w2[s]
  for (int s = 0; s < NSH; ++s) {
    gemm_swiglu<<<dim3(SH_HID / 64, TOK / 128), 256, 0, stream>>>(
        X16, sh_w1 + (size_t)s * DIMX * SH_HID * 2, Hbuf,
        nullptr, nullptr, 0, TOK, DIMX, SH_HID);
    gemm_acc<<<dim3(DIMX / 128, TOK / 128), 256, 0, stream>>>(
        Hbuf, sh_w2 + (size_t)s * SH_HID * DIMX, out,
        nullptr, nullptr, nullptr, nullptr, 0, TOK, SH_HID, DIMX, 0.5f);
  }

  // 6) routed experts: only tokens routed to e (top-2), gate-weighted scatter-add
  for (int e = 0; e < NEXP; ++e) {
    gemm_swiglu<<<dim3(EXP_HID / 64, TOK / 128), 256, 0, stream>>>(
        X16, exp_w1 + (size_t)e * DIMX * EXP_HID * 2, Hbuf,
        toklist + (size_t)e * TOK, cnt, e, TOK, DIMX, EXP_HID);
    gemm_mid<<<dim3(DIMX / 128, TOK / 128), 256, 0, stream>>>(
        Hbuf, exp_w2 + (size_t)e * EXP_HID * DIMX, MID, cnt, e, EXP_HID, DIMX);
    gemm_acc<<<dim3(DIMX / 128, TOK / 128), 256, 0, stream>>>(
        MID, exp_wo + (size_t)e * DIMX * DIMX, out,
        toklist + (size_t)e * TOK, gatelist + (size_t)e * TOK,
        exp_bo + (size_t)e * DIMX, cnt, e, TOK, DIMX, DIMX, 1.0f);
  }
}
